// GroupWiseDeformableConv_38087769981585
// MI455X (gfx1250) — matrix operand order repaired and checked by reading, not compiled, not run on hardware
//
#include <hip/hip_runtime.h>
#include <hip/hip_bf16.h>

// ---------------------------------------------------------------------------
// GroupWiseDeformableConv for MI455X (gfx1250, wave32, WMMA)
//   x      [16,256,64,64] f32
//   off_w  [4,18,64,3,3]  f32   off_b [4,18] f32
//   def_w  [4,64,64,3,3]  f32   def_b [4,64] f32
//   out    [16,256,64,64] f32
// Deformable conv einsum is a GEMM M=64, K=576, N=65536 per group ->
// fused bilinear-sample + v_wmma_f32_16x16x32_bf16 with f32 accumulation.
// ---------------------------------------------------------------------------

#define KG   4
#define CH   256
#define CG   64
#define KK   9
#define B_   16
#define H_   64
#define W_   64
#define HW   (H_ * W_)

typedef __attribute__((ext_vector_type(16))) __bf16 v16bf;
typedef __attribute__((ext_vector_type(8)))  float  v8f;

__device__ __forceinline__ unsigned short f2bf(float f) {
  unsigned int u = __float_as_uint(f);
  unsigned int r = u + 0x7FFFu + ((u >> 16) & 1u);   // round-to-nearest-even
  return (unsigned short)(r >> 16);
}

// ---------------- Stage 0: repack def_w -> bf16 [g][tap][o][i] -------------
__global__ void __launch_bounds__(256)
wprep_kernel(const float* __restrict__ def_w, unsigned short* __restrict__ w_bf) {
  int gid = blockIdx.x * 256 + threadIdx.x;           // KG*KK*CG*CG = 147456
  if (gid >= KG * KK * CG * CG) return;
  int i = gid & 63;
  int o = (gid >> 6) & 63;
  int r = gid >> 12;
  int k = r % KK;
  int g = r / KK;
  float v = def_w[(((g * CG + o) * CG + i) * KK) + k];
  w_bf[gid] = f2bf(v);
}

// ---------------- Stage 1: offset conv (Cg -> 18, 3x3, pad 1) --------------
__global__ void __launch_bounds__(256)
offconv_kernel(const float* __restrict__ x, const float* __restrict__ off_w,
               const float* __restrict__ off_b, float* __restrict__ offs) {
  int gid = blockIdx.x * 256 + threadIdx.x;           // KG*B*18*HW
  int sp = gid & (HW - 1);
  int t  = gid >> 12;
  int oc = t % 18; t /= 18;
  int b  = t % B_;
  int g  = t / B_;
  int y  = sp >> 6;
  int xx = sp & 63;

  float acc = off_b[g * 18 + oc];
  const float* wb = off_w + (size_t)(g * 18 + oc) * CG * KK;
  const float* xb = x + (size_t)(b * CH + g * CG) * HW;
#pragma unroll 4
  for (int i = 0; i < CG; ++i) {
    const float* w9 = wb + i * KK;
    const float* ch = xb + (size_t)i * HW;
#pragma unroll
    for (int dy = -1; dy <= 1; ++dy) {
#pragma unroll
      for (int dx = -1; dx <= 1; ++dx) {
        int yy = y + dy, xc = xx + dx;
        if (yy >= 0 && yy < H_ && xc >= 0 && xc < W_)
          acc += ch[yy * W_ + xc] * w9[(dy + 1) * 3 + (dx + 1)];
      }
    }
  }
  offs[gid] = acc;
}

// ------- Stage 2: fused bilinear sampling + WMMA GEMM + bias + store -------
// grid (y=64, b=16, g=4), 256 threads (8 waves).
__global__ void __launch_bounds__(256)
deform_wmma_kernel(const float* __restrict__ x, const float* __restrict__ offs,
                   const unsigned short* __restrict__ w_bf,
                   const float* __restrict__ def_b, float* __restrict__ out) {
  __shared__ __align__(16) unsigned short As[CG * CG];   // [m][i] bf16, 8KB
  __shared__ __align__(16) unsigned short Bs[W_ * CG];   // [n][i] bf16, 8KB

  const int tid  = threadIdx.x;
  const int lane = tid & 31;
  const int wave = tid >> 5;
  const int y = blockIdx.x, b = blockIdx.y, g = blockIdx.z;

  const int mt  = wave & 3;          // M-tile 0..3
  const int ntp = (wave >> 2) * 2;   // N-tile pair base: 0 or 2

  const size_t xbase   = (size_t)(b * CH + g * CG) * HW;
  const size_t offbase = (size_t)((g * B_ + b) * 18) * HW + (size_t)y * W_;

  const int p     = tid & 63;        // my pixel (invariant across gather iters)
  const int cbase = tid >> 6;        // channel start 0..3, step 4

  v8f acc0 = {}, acc1 = {};

  for (int k = 0; k < KK; ++k) {
    // ---- per-thread bilinear coords for pixel p (computed 4x redundantly,
    //      all reads L2-resident; avoids an extra barrier + EXEC churn) ----
    float dy = offs[offbase + (size_t)(2 * k) * HW + p];
    float dx = offs[offbase + (size_t)(2 * k + 1) * HW + p];
    float py = dy + (float)(k / 3 - 1) + (float)y;
    float px = dx + (float)(k % 3 - 1) + (float)p;
    float fy = floorf(py), fx = floorf(px);
    int   y0 = (int)fy,  x0 = (int)fx;
    float wy = py - fy,  wx = px - fx;

    // corner weights with zero-padding folded in (w=0 outside the image)
    float vy0 = (y0 >= 0 && y0 < H_) ? 1.0f : 0.0f;
    float vy1 = (y0 + 1 >= 0 && y0 + 1 < H_) ? 1.0f : 0.0f;
    float vx0 = (x0 >= 0 && x0 < W_) ? 1.0f : 0.0f;
    float vx1 = (x0 + 1 >= 0 && x0 + 1 < W_) ? 1.0f : 0.0f;
    float w00 = (1.0f - wy) * (1.0f - wx) * vy0 * vx0;
    float w01 = (1.0f - wy) * wx          * vy0 * vx1;
    float w10 = wy * (1.0f - wx)          * vy1 * vx0;
    float w11 = wy * wx                   * vy1 * vx1;

    int yc0 = min(max(y0, 0), H_ - 1);
    int yc1 = min(max(y0 + 1, 0), H_ - 1);
    int xc0 = min(max(x0, 0), W_ - 1);
    int xc1 = min(max(x0 + 1, 0), W_ - 1);
    int o00 = yc0 * W_ + xc0, o01 = yc0 * W_ + xc1;
    int o10 = yc1 * W_ + xc0, o11 = yc1 * W_ + xc1;

    // ---- stage bf16 weight chunk As[m][i] for this tap ----
    {
      const unsigned int* src =
          (const unsigned int*)(w_bf + (size_t)(g * KK + k) * CG * CG);
      unsigned int* dst = (unsigned int*)As;
#pragma unroll
      for (int it = 0; it < 8; ++it) dst[tid + it * 256] = src[tid + it * 256];
    }
    // ---- prefetch next tap's weight chunk + offset rows (global_prefetch)
    if (k + 1 < KK) {
      __builtin_prefetch(w_bf + (size_t)(g * KK + k + 1) * CG * CG + tid * 8, 0, 3);
      __builtin_prefetch(&offs[offbase + (size_t)(2 * k + 2) * HW + p], 0, 3);
      __builtin_prefetch(&offs[offbase + (size_t)(2 * k + 3) * HW + p], 0, 3);
    }

    // ---- bilinear gather -> Bs[p][c], 4 loads + 4 FMA per element ----
    const float* chan0 = x + xbase + (size_t)cbase * HW;
#pragma unroll 4
    for (int it = 0; it < 16; ++it) {
      const float* ch = chan0 + (size_t)(it * 4) * HW;    // c = cbase + it*4
      float s = ch[o00] * w00 + ch[o01] * w01 + ch[o10] * w10 + ch[o11] * w11;
      Bs[p * CG + (cbase + it * 4)] = f2bf(s);
    }
    __syncthreads();

    // ---- WMMA over this K chunk (i = 0..63 -> 2 halves of 32) ----
    // A frag: lane = M (mod 16); lanes<16 hold K {kb..kb+15},
    //         kb = 32h; lanes>=16 hold K {kb+16..kb+31}.
    // B frag: lane = N (mod 16); lanes<16 hold K {kb..kb+15},
    //         lanes>=16 hold K {kb+16..kb+31}.
#pragma unroll
    for (int h = 0; h < 2; ++h) {
      v16bf af, bf0, bf1;
      {
        int m  = mt * 16 + (lane & 15);
        int kb = h * 32 + ((lane < 16) ? 0 : 16);
        union { uint4 q[2]; v16bf v; } u;
        u.q[0] = *(const uint4*)(As + m * CG + kb);
        u.q[1] = *(const uint4*)(As + m * CG + kb + 8);
        af = u.v;
      }
      {
        int n  = ntp * 16 + (lane & 15);
        int kb = h * 32 + ((lane < 16) ? 0 : 16);
        union { uint4 q[2]; v16bf v; } u;
        u.q[0] = *(const uint4*)(Bs + n * CG + kb);
        u.q[1] = *(const uint4*)(Bs + n * CG + kb + 8);
        bf0 = u.v;
      }
      {
        int n  = (ntp + 1) * 16 + (lane & 15);
        int kb = h * 32 + ((lane < 16) ? 0 : 16);
        union { uint4 q[2]; v16bf v; } u;
        u.q[0] = *(const uint4*)(Bs + n * CG + kb);
        u.q[1] = *(const uint4*)(Bs + n * CG + kb + 8);
        bf1 = u.v;
      }
      acc0 = __builtin_amdgcn_wmma_f32_16x16x32_bf16(
          false, af, false, bf0, (short)0, acc0, false, false);
      acc1 = __builtin_amdgcn_wmma_f32_16x16x32_bf16(
          false, af, false, bf1, (short)0, acc1, false, false);
    }
    __syncthreads();   // protect LDS before next tap rewrites it
  }

  // ---- bias + store (C layout: VGPR r -> M = r + 8*(lane>=16), N = lane&15)
  const int n    = lane & 15;
  const int mofs = (lane < 16) ? 0 : 8;
  const size_t outrow = (size_t)(b * CH + g * CG) * HW + (size_t)y * W_;
#pragma unroll
  for (int r = 0; r < 8; ++r) {
    int m = mt * 16 + r + mofs;
    float bias = def_b[g * CG + m];
    out[outrow + (size_t)m * HW + ntp * 16 + n]       = acc0[r] + bias;
    out[outrow + (size_t)m * HW + (ntp + 1) * 16 + n] = acc1[r] + bias;
  }
}

// ---------------------------------------------------------------------------
extern "C" void kernel_launch(void* const* d_in, const int* in_sizes, int n_in,
                              void* d_out, int out_size, void* d_ws, size_t ws_size,
                              hipStream_t stream) {
  const float* x     = (const float*)d_in[0];
  const float* off_w = (const float*)d_in[1];
  const float* off_b = (const float*)d_in[2];
  const float* def_w = (const float*)d_in[3];
  const float* def_b = (const float*)d_in[4];
  float* out = (float*)d_out;

  // workspace: offsets f32 [4][16][18][4096], then bf16 weights [4][9][64][64]
  const size_t offs_elems = (size_t)KG * B_ * 18 * HW;       // 4,718,592
  float*          offs = (float*)d_ws;
  unsigned short* w_bf = (unsigned short*)((char*)d_ws + offs_elems * sizeof(float));

  {
    int n = KG * KK * CG * CG;                               // 147,456
    wprep_kernel<<<(n + 255) / 256, 256, 0, stream>>>(def_w, w_bf);
  }
  {
    int n = (int)offs_elems;
    offconv_kernel<<<(n + 255) / 256, 256, 0, stream>>>(x, off_w, off_b, offs);
  }
  {
    dim3 grid(H_, B_, KG);                                   // 64 x 16 x 4
    deform_wmma_kernel<<<grid, 256, 0, stream>>>(x, offs, w_bf, def_b, out);
  }
}
